// CausalSelfAttention_23373212025283
// MI455X (gfx1250) — compile-verified, matrix operand order checked by reading, not run on hardware
//
#include <hip/hip_runtime.h>
#include <hip/hip_bf16.h>

// Problem constants (from reference)
#define B_  2
#define T_  2048
#define C_  1024
#define H_  16
#define DH_ 64
#define BT_ (B_ * T_)   // 4096

typedef _Float16 v16h  __attribute__((ext_vector_type(16)));
typedef _Float16 v8h   __attribute__((ext_vector_type(8)));
typedef float    v8f   __attribute__((ext_vector_type(8)));
typedef unsigned int u32x4 __attribute__((ext_vector_type(4)));
typedef int      i32x4 __attribute__((ext_vector_type(4)));
typedef int      i32x8 __attribute__((ext_vector_type(8)));

union Frag16 { v16h v; v8h h[2]; };

static __device__ __forceinline__ void lds_fence() {
    // Prevent compiler reordering of LDS phases; hardware keeps same-wave
    // LDS ops in order (DScnt in-order).
    asm volatile("" ::: "memory");
}

// ---------------------------------------------------------------------------
// Tensor Data Mover: 2-D tile (elements are 2 bytes) from global -> LDS.
//   tile_d0 = contiguous elements per row, tile_d1 = rows,
//   stride  = elements between rows in global memory.
// D# layout per CDNA5 ISA ch.8 (group0: flags/lds/global/type, group1: dims).
// ---------------------------------------------------------------------------
#if __has_builtin(__builtin_amdgcn_tensor_load_to_lds)
#define HAVE_TDM 1
#pragma message("CDNA5 probe: TDM path ENABLED (__builtin_amdgcn_tensor_load_to_lds present)")
#if __clang_major__ >= 23
#pragma message("CDNA5 probe: TDM builtin arity = 6 (clang>=23)")
#else
#pragma message("CDNA5 probe: TDM builtin arity = 5 (clang<23)")
#endif
static __device__ __forceinline__ void tdm_load_2d_f16(
    const _Float16* __restrict__ gsrc, void* ldsDst,
    unsigned tile_d0, unsigned tile_d1, unsigned long long stride_elems)
{
    unsigned long long ga = (unsigned long long)(uintptr_t)gsrc;
    unsigned lds = (unsigned)(uintptr_t)ldsDst;  // flat LDS addr low 32 = offset

    u32x4 g0;
    g0[0] = 1u;                                   // count=1, user descriptor
    g0[1] = lds;                                  // lds_addr (bytes)
    g0[2] = (unsigned)ga;                         // global_addr[31:0]
    g0[3] = (unsigned)((ga >> 32) & 0x01FFFFFFu)  // global_addr[56:32]
          | (2u << 30);                           // type = 2 ("image")

    const unsigned TD0 = 0x40000000u;             // tensor_dim0 (huge, no OOB clip)
    const unsigned TD1 = 0x40000000u;             // tensor_dim1
    i32x8 g1;
    g1[0] = 0x00010000;                                   // data_size=1 (2B)
    g1[1] = (int)((TD0 & 0xFFFFu) << 16);                 // tensor_dim0 lo16
    g1[2] = (int)(((TD0 >> 16) & 0xFFFFu)
          |       ((TD1 & 0xFFFFu) << 16));               // td0 hi | td1 lo
    g1[3] = (int)(((TD1 >> 16) & 0xFFFFu)
          |       ((tile_d0 & 0xFFFFu) << 16));           // td1 hi | tile_dim0
    g1[4] = (int)(tile_d1 & 0xFFFFu);                     // tile_dim1 (tile_dim2=0)
    g1[5] = (int)(unsigned)(stride_elems & 0xFFFFFFFFull);// dim0_stride lo32
    g1[6] = (int)(unsigned)((stride_elems >> 32) & 0xFFFFull); // dim0_stride hi16
    g1[7] = 0;

    i32x4 z4 = (i32x4)0;
#if __clang_major__ >= 23
    i32x8 z8 = (i32x8)0;
    __builtin_amdgcn_tensor_load_to_lds(g0, g1, z4, z4, z8, 0);
#else
    __builtin_amdgcn_tensor_load_to_lds(g0, g1, z4, z4, 0);
#endif
}
#else
#define HAVE_TDM 0
#pragma message("CDNA5 probe: TDM builtin ABSENT - using cooperative-copy fallback")
#endif

// ---------------------------------------------------------------------------
// Kernel 1: fused QKV projection GEMM.
//   out = x[4096,1024] @ W[1024,1024] + bias, f16 head-major output.
//   grid.z in {0,1,2} selects Q, K, V; V written transposed [B,H,DH,T].
// Block: 256 threads = 8 waves; block tile M=128, N=128, K-step 32.
// Wave (wm 0..3, wn 0..1) computes 32x64 = 2x4 WMMA fragments (8 wmma/iter).
// ---------------------------------------------------------------------------
__global__ __launch_bounds__(256)
void qkv_gemm(const float* __restrict__ x,
              const float* __restrict__ Wq, const float* __restrict__ bq,
              const float* __restrict__ Wk, const float* __restrict__ bk,
              const float* __restrict__ Wv, const float* __restrict__ bv,
              _Float16* __restrict__ Qh, _Float16* __restrict__ Kh,
              _Float16* __restrict__ Vt)
{
    const int mat = blockIdx.z;
    const float* W    = (mat == 0) ? Wq : (mat == 1) ? Wk : Wv;
    const float* bias = (mat == 0) ? bq : (mat == 1) ? bk : bv;
    _Float16*    dst  = (mat == 0) ? Qh : (mat == 1) ? Kh : Vt;

    const int m0 = blockIdx.x * 128;
    const int n0 = blockIdx.y * 128;

    __shared__ __align__(16) _Float16 sA[128][40];  // x tile   (M x K)
    __shared__ __align__(16) _Float16 sB[128][40];  // W tile^T (N x K)

    const int tid   = threadIdx.x;
    const int lane  = tid & 31;
    const int w     = tid >> 5;          // 0..7
    const int wm    = w >> 1;            // 0..3
    const int wn    = w & 1;             // 0..1
    const int lr    = lane & 15;
    const int half8 = (lane >> 4) << 3;  // 0 or 8

    v8f acc[2][4] = {};

    for (int k0 = 0; k0 < C_; k0 += 32) {
        __syncthreads();
        // A tile: 128x32 f32 -> f16 (1024 float4 over 256 threads)
        #pragma unroll
        for (int i = 0; i < 4; ++i) {
            int idx = tid + i * 256;
            int r   = idx >> 3;
            int c4  = (idx & 7) << 2;
            float4 a = *(const float4*)(x + (size_t)(m0 + r) * C_ + k0 + c4);
            sA[r][c4 + 0] = (_Float16)a.x;
            sA[r][c4 + 1] = (_Float16)a.y;
            sA[r][c4 + 2] = (_Float16)a.z;
            sA[r][c4 + 3] = (_Float16)a.w;
        }
        // B tile: W[k0..k0+32, n0..n0+128] transposed into sB[n][k]
        #pragma unroll
        for (int i = 0; i < 4; ++i) {
            int idx = tid + i * 256;           // 0..1023
            int kr  = idx >> 5;                // 32 float4 per K row
            int c4  = (idx & 31) << 2;
            float4 bb = *(const float4*)(W + (size_t)(k0 + kr) * C_ + n0 + c4);
            sB[c4 + 0][kr] = (_Float16)bb.x;
            sB[c4 + 1][kr] = (_Float16)bb.y;
            sB[c4 + 2][kr] = (_Float16)bb.z;
            sB[c4 + 3][kr] = (_Float16)bb.w;
        }
        __syncthreads();

        Frag16 af[2], bf[4];
        #pragma unroll
        for (int i = 0; i < 2; ++i) {
            const _Float16* p = &sA[wm * 32 + i * 16 + lr][half8];
            af[i].h[0] = *(const v8h*)(p);
            af[i].h[1] = *(const v8h*)(p + 16);
        }
        #pragma unroll
        for (int j = 0; j < 4; ++j) {
            const _Float16* p = &sB[wn * 64 + j * 16 + lr][half8];
            bf[j].h[0] = *(const v8h*)(p);
            bf[j].h[1] = *(const v8h*)(p + 16);
        }
        #pragma unroll
        for (int i = 0; i < 2; ++i)
            #pragma unroll
            for (int j = 0; j < 4; ++j)
                acc[i][j] = __builtin_amdgcn_wmma_f32_16x16x32_f16(
                    false, af[i].v, false, bf[j].v, (short)0, acc[i][j],
                    false, false);
    }

    // epilogue: bias, convert, scatter head-major (V transposed)
    #pragma unroll
    for (int i = 0; i < 2; ++i) {
        #pragma unroll
        for (int j = 0; j < 4; ++j) {
            const int n  = n0 + wn * 64 + j * 16 + lr;
            const float bval = bias[n];
            const int hh = n >> 6;
            const int d  = n & 63;
            #pragma unroll
            for (int r = 0; r < 8; ++r) {
                int m = m0 + wm * 32 + i * 16 + r + half8;
                int b = m >> 11;
                int t = m & (T_ - 1);
                float val = acc[i][j][r] + bval;
                size_t idx = (mat == 2)
                    ? ((size_t)((b * H_ + hh) * DH_ + d) * T_ + t)   // V^T
                    : ((size_t)((b * H_ + hh) * T_ + t) * DH_ + d);  // Q, K
                dst[idx] = (_Float16)val;
            }
        }
    }
}

// ---------------------------------------------------------------------------
// Kernel 2: flash attention with TDM-fed, double-buffered K/V LDS tiles.
// Block: 128 threads = 4 waves, 64 queries of one (b,h); 32 keys per step.
// Wave 0 drives the Tensor Data Mover; waves consume shared LDS tiles.
// ---------------------------------------------------------------------------
__global__ __launch_bounds__(128)
void attn(const _Float16* __restrict__ Qh, const _Float16* __restrict__ Kh,
          const _Float16* __restrict__ Vt, const int* __restrict__ mask,
          float* __restrict__ out)
{
    const int bh    = blockIdx.x;
    const int b     = bh >> 4;
    const int h     = bh & 15;
    const int q0    = blockIdx.y * 64;
    const int w     = threadIdx.x >> 5;
    const int lane  = threadIdx.x & 31;
    const int lr    = lane & 15;
    const int half8 = (lane >> 4) << 3;

    const int qbase = q0 + w * 16;

    __shared__ __align__(16) _Float16 sK[2][32][64];   // keys x features
    __shared__ __align__(16) _Float16 sV[2][64][32];   // features x keys
    __shared__ float sS[4][16][33];
    __shared__ __align__(16) _Float16 sP[4][16][40];
    __shared__ float sM[4][16], sL[4][16], sAl[4][16];
    __shared__ int   sKm[4][32];
    __shared__ int   sQm[4][16];

    const size_t headQK = (size_t)(b * H_ + h) * T_ * DH_;
    const size_t headV  = (size_t)(b * H_ + h) * DH_ * T_;
    const _Float16* Kbase = Kh + headQK;
    const _Float16* Vbase = Vt + headV;

    // Q fragments (two K=32 steps over DH=64), loaded once from global.
    Frag16 aq[2];
    {
        const _Float16* qrow = Qh + headQK + (size_t)(qbase + lr) * DH_;
        #pragma unroll
        for (int s = 0; s < 2; ++s) {
            aq[s].h[0] = *(const v8h*)(qrow + s * 32 + half8);
            aq[s].h[1] = *(const v8h*)(qrow + s * 32 + half8 + 16);
        }
    }

    if (lane < 16) {
        sM[w][lane]  = -1e30f;
        sL[w][lane]  = 0.0f;
        sQm[w][lane] = mask[b * T_ + qbase + lane];
    }
    lds_fence();

    // ---- tile producer: TDM if available, else cooperative copy ----
#if HAVE_TDM
    #define LOAD_TILES(buf, kk)                                              \
        do { if (w == 0) {                                                   \
            tdm_load_2d_f16(Kbase + (size_t)(kk) * DH_, &sK[buf][0][0],      \
                            32 * DH_, 1, 32 * DH_);     /* contiguous 4KB */ \
            tdm_load_2d_f16(Vbase + (kk), &sV[buf][0][0],                    \
                            32, DH_, (unsigned long long)T_);                \
        } } while (0)
    #define WAIT_TILES() \
        do { if (w == 0) __builtin_amdgcn_s_wait_tensorcnt(0); } while (0)
#else
    #define LOAD_TILES(buf, kk)                                              \
        do {                                                                 \
            /* K: 256 x 16B chunks; V: 64 rows x 2 chunks -> 128 chunks */   \
            for (int c = threadIdx.x; c < 256; c += 128)                     \
                *((v8h*)&sK[buf][0][0] + c) =                                \
                    *((const v8h*)(Kbase + (size_t)(kk) * DH_) + c);         \
            for (int c = threadIdx.x; c < 128; c += 128) {                   \
                int vr = c >> 1, vc = (c & 1) << 3;                          \
                *(v8h*)&sV[buf][vr][vc] =                                    \
                    *(const v8h*)(Vbase + (size_t)vr * T_ + (kk) + vc);      \
            }                                                                \
        } while (0)
    #define WAIT_TILES() do { } while (0)
#endif

    // prologue: fill buffer 0
    LOAD_TILES(0, 0);
    WAIT_TILES();
    __syncthreads();

    v8f o[4] = {};
    const float scale = 0.125f;           // 1/sqrt(DH)
    const int   nIter = T_ / 32;

    for (int it = 0; it < nIter; ++it) {
        const int p  = it & 1;
        const int k0 = it * 32;

        // kick off next tile into the other buffer (overlaps with compute)
        if (it + 1 < nIter) LOAD_TILES(p ^ 1, k0 + 32);

        // key mask for this tile (per wave; same-wave LDS is in-order)
        sKm[w][lane] = mask[b * T_ + k0 + lane];

        // ---- S = Q @ K^T : 16x32, two N-subtiles from shared K tile ----
        #pragma unroll
        for (int j = 0; j < 2; ++j) {
            v8f accs = {};
            #pragma unroll
            for (int s = 0; s < 2; ++s) {
                Frag16 bk;
                const _Float16* krow = &sK[p][j * 16 + lr][0];
                bk.h[0] = *(const v8h*)(krow + s * 32 + half8);
                bk.h[1] = *(const v8h*)(krow + s * 32 + half8 + 16);
                accs = __builtin_amdgcn_wmma_f32_16x16x32_f16(
                    false, aq[s].v, false, bk.v, (short)0, accs, false, false);
            }
            #pragma unroll
            for (int r = 0; r < 8; ++r)
                sS[w][r + half8][j * 16 + lr] = accs[r];
        }
        lds_fence();

        // ---- online softmax: lanes 0..15 own one query row each ----
        if (lane < 16) {
            float mo = sM[w][lane], lo = sL[w][lane];
            float tm = -1e30f;
            #pragma unroll 8
            for (int j = 0; j < 32; ++j)
                if (sKm[w][j]) tm = fmaxf(tm, sS[w][lane][j] * scale);
            float nm = fmaxf(mo, tm);
            float al = __expf(mo - nm);
            float ps = 0.0f;
            #pragma unroll 8
            for (int j = 0; j < 32; ++j) {
                float pv = sKm[w][j] ? __expf(sS[w][lane][j] * scale - nm) : 0.0f;
                sP[w][lane][j] = (_Float16)pv;
                ps += pv;
            }
            sM[w][lane]  = nm;
            sL[w][lane]  = al * lo + ps;
            sAl[w][lane] = al;
        }
        lds_fence();

        // ---- rescale running accumulators ----
        float ar[8];
        #pragma unroll
        for (int r = 0; r < 8; ++r) ar[r] = sAl[w][r + half8];
        #pragma unroll
        for (int f = 0; f < 4; ++f)
            #pragma unroll
            for (int r = 0; r < 8; ++r) o[f][r] *= ar[r];

        // ---- P fragment (A 16x32 layout) ----
        Frag16 ap;
        {
            const _Float16* prow = &sP[w][lr][0];
            ap.h[0] = *(const v8h*)(prow + half8);
            ap.h[1] = *(const v8h*)(prow + half8 + 16);
        }

        // ---- O += P @ V : four 16-feature N-subtiles from shared V tile ----
        #pragma unroll
        for (int f = 0; f < 4; ++f) {
            Frag16 bv;
            const _Float16* vrow = &sV[p][f * 16 + lr][0];
            bv.h[0] = *(const v8h*)(vrow + half8);
            bv.h[1] = *(const v8h*)(vrow + half8 + 16);
            o[f] = __builtin_amdgcn_wmma_f32_16x16x32_f16(
                false, ap.v, false, bv.v, (short)0, o[f], false, false);
        }

        // next tile must be resident before anyone touches buffer p^1
        WAIT_TILES();
        __syncthreads();
    }

    // ---- finalize: divide by l, zero masked query rows ----
    #pragma unroll
    for (int f = 0; f < 4; ++f) {
        const int c = h * DH_ + f * 16 + lr;
        #pragma unroll
        for (int r = 0; r < 8; ++r) {
            int   row   = r + half8;
            int   q     = qbase + row;
            float denom = sL[w][row];
            int   qm    = sQm[w][row];
            float val   = (qm && denom > 0.0f) ? o[f][r] / denom : 0.0f;
            out[(size_t)(b * T_ + q) * C_ + c] = val;
        }
    }
}

// ---------------------------------------------------------------------------
extern "C" void kernel_launch(void* const* d_in, const int* in_sizes, int n_in,
                              void* d_out, int out_size, void* d_ws, size_t ws_size,
                              hipStream_t stream) {
    const float* x    = (const float*)d_in[0];
    const float* Wq   = (const float*)d_in[1];
    const float* bq   = (const float*)d_in[2];
    const float* Wk   = (const float*)d_in[3];
    const float* bk   = (const float*)d_in[4];
    const float* Wv   = (const float*)d_in[5];
    const float* bv   = (const float*)d_in[6];
    const int*   mask = (const int*)d_in[7];
    float*       out  = (float*)d_out;

    const size_t headElems = (size_t)B_ * H_ * T_ * DH_;  // 4M halves = 8 MB
    _Float16* Qh = (_Float16*)d_ws;
    _Float16* Kh = Qh + headElems;
    _Float16* Vt = Kh + headElems;

    qkv_gemm<<<dim3(BT_ / 128, C_ / 128, 3), 256, 0, stream>>>(
        x, Wq, bq, Wk, bk, Wv, bv, Qh, Kh, Vt);
    attn<<<dim3(B_ * H_, T_ / 64), 128, 0, stream>>>(Qh, Kh, Vt, mask, out);
}
